// Phi3Mamba_24704651887233
// MI455X (gfx1250) — matrix operand order repaired and checked by reading, compiled, not run on hardware
//
#include <hip/hip_runtime.h>

// ---------------------------------------------------------------------------
// Mamba forward for MI455X (gfx1250, wave32, WMMA, async global->LDS).
// ---------------------------------------------------------------------------

typedef __bf16 bf16;
typedef __attribute__((ext_vector_type(16))) __bf16 v16bf;
typedef __attribute__((ext_vector_type(8)))  __bf16 v8bf;
typedef __attribute__((ext_vector_type(8)))  float  v8f;
typedef __attribute__((ext_vector_type(4)))  int    v4i;

#define B_SZ    2
#define SEQ     4096
#define D_MODEL 2048
#define D_INNER 4096
#define DT_RANK 128
#define D_STATE 16
#define BLM     (B_SZ * SEQ)            /* 8192 flattened (b,l) rows */
#define SSM_N   (DT_RANK + 2 * D_STATE) /* 160 */
#define SEGS    64                      /* scan segments per sequence */
#define SEGLEN  (SEQ / SEGS)            /* 64 steps per segment */

// ---------------------------------------------------------------------------
// Async global->LDS helpers (gfx1250 GLOBAL_LOAD_ASYNC_TO_LDS_B128, ASYNCcnt).
// Builtin signature (from hipcc diagnostic): (v4i AS1*, v4i AS3*, Ii, Ii).
// ---------------------------------------------------------------------------
__device__ __forceinline__ void async_b128(const bf16* g, bf16* l) {
#if __has_builtin(__builtin_amdgcn_global_load_async_to_lds_b128)
  __builtin_amdgcn_global_load_async_to_lds_b128(
      (__attribute__((address_space(1))) v4i*)(void*)g,
      (__attribute__((address_space(3))) v4i*)(void*)l,
      /*imm offset=*/0, /*cpol=*/0);
#else
  *(v8bf*)l = *(const v8bf*)g;   // sync fallback: load + ds_store
#endif
}

__device__ __forceinline__ void wait_async0() {
#if __has_builtin(__builtin_amdgcn_s_wait_asynccnt)
  __builtin_amdgcn_s_wait_asynccnt(0);
#else
  asm volatile("s_wait_asynccnt 0" ::: "memory");
#endif
}

// ---------------------------------------------------------------------------
// f32 -> bf16 cast
// ---------------------------------------------------------------------------
__global__ void k_cast_bf16(const float* __restrict__ src,
                            bf16* __restrict__ dst, size_t n) {
  size_t i = (size_t)blockIdx.x * blockDim.x + threadIdx.x;
  if (i < n) dst[i] = (bf16)src[i];
}

// ---------------------------------------------------------------------------
// GEMM: C[M,N] = A[M,K] * B[N,K]^T    (einsum('bld,ed->ble') pattern)
// A,B bf16 row-major; C f32. Block tile 128x128x64, 256 threads = 8 waves,
// each wave owns 32x64 => 2x4 WMMA tiles, 16 v_wmma per K-tile.
// ---------------------------------------------------------------------------
__global__ __launch_bounds__(256)
void k_gemm_bf16_wmma(const bf16* __restrict__ A,
                      const bf16* __restrict__ B,
                      float* __restrict__ C,
                      int M, int N, int K) {
  constexpr int BM = 128, BN = 128, BK = 64;
  constexpr int LDA = 72;   // 144B rows: 16B-aligned subtiles, bank-skewed
  constexpr int LDB = 80;   // 160B rows: 32B-aligned v16bf reads
  __shared__ __align__(32) bf16 As[BM * LDA];
  __shared__ __align__(32) bf16 Bs[BN * LDB];

  const int tid  = threadIdx.x;
  const int lane = tid & 31;
  const int wid  = tid >> 5;
  const int lm   = lane & 15;       // index within 16-wide tile
  const int lhi  = lane >> 4;       // half-wave selector per ISA layout
  const int wm   = (wid & 3) * 32;  // wave row offset in block tile
  const int wn   = (wid >> 2) * 64; // wave col offset in block tile
  const int bm   = blockIdx.y * BM;
  const int bn   = blockIdx.x * BN;

  // staging: thread t covers tile row t>>1, K-halves of 32 elems
  const int lr = tid >> 1;
  const int lc = (tid & 1) * 32;
  const bool bn_full = (bn + BN) <= N;   // uniform scalar branch

  v8f acc[2][4] = {};

  for (int k0 = 0; k0 < K; k0 += BK) {
    // ---- stage A (always full in M/K) via async direct-to-LDS ----
    {
      const bf16* ga = A + (size_t)(bm + lr) * K + k0 + lc;
      async_b128(ga,      &As[lr * LDA + lc]);
      async_b128(ga + 8,  &As[lr * LDA + lc + 8]);
      async_b128(ga + 16, &As[lr * LDA + lc + 16]);
      async_b128(ga + 24, &As[lr * LDA + lc + 24]);
    }
    // ---- stage B ----
    if (bn_full) {
      const bf16* gb = B + (size_t)(bn + lr) * K + k0 + lc;
      async_b128(gb,      &Bs[lr * LDB + lc]);
      async_b128(gb + 8,  &Bs[lr * LDB + lc + 8]);
      async_b128(gb + 16, &Bs[lr * LDB + lc + 16]);
      async_b128(gb + 24, &Bs[lr * LDB + lc + 24]);
    } else {
      v8bf b0 = {}, b1 = {}, b2 = {}, b3 = {};
      const int gn = bn + lr;
      if (gn < N) {
        const bf16* gb = B + (size_t)gn * K + k0 + lc;
        b0 = *(const v8bf*)gb;
        b1 = *(const v8bf*)(gb + 8);
        b2 = *(const v8bf*)(gb + 16);
        b3 = *(const v8bf*)(gb + 24);
      }
      *(v8bf*)&Bs[lr * LDB + lc]      = b0;
      *(v8bf*)&Bs[lr * LDB + lc + 8]  = b1;
      *(v8bf*)&Bs[lr * LDB + lc + 16] = b2;
      *(v8bf*)&Bs[lr * LDB + lc + 24] = b3;
    }
    wait_async0();
    __syncthreads();

    // ---- two 32-deep K-steps of WMMA per staged tile ----
#pragma unroll
    for (int ks = 0; ks < 2; ++ks) {
      const int kb = ks * 32;
      // A 16x32 bf16 frag: lane(m=lm,hi=lhi) holds K = kb+hi*16 .. +15 contiguous
      v16bf af[2];
#pragma unroll
      for (int mt = 0; mt < 2; ++mt) {
        const int m = wm + mt * 16 + lm;
        v8bf lo = *(const v8bf*)&As[m * LDA + kb + lhi * 16];
        v8bf hi = *(const v8bf*)&As[m * LDA + kb + lhi * 16 + 8];
        af[mt] = __builtin_shufflevector(lo, hi,
                   0, 1, 2, 3, 4, 5, 6, 7, 8, 9, 10, 11, 12, 13, 14, 15);
      }
      // B 32x16 bf16 frag: lane(n=lm,hi=lhi) holds K = kb+hi*16 .. +15 contiguous
      v16bf bfrag[4];
#pragma unroll
      for (int nt = 0; nt < 4; ++nt) {
        const int n = wn + nt * 16 + lm;
        bfrag[nt] = *(const v16bf*)&Bs[n * LDB + kb + lhi * 16];
      }
#pragma unroll
      for (int mt = 0; mt < 2; ++mt)
#pragma unroll
        for (int nt = 0; nt < 4; ++nt)
          acc[mt][nt] = __builtin_amdgcn_wmma_f32_16x16x32_bf16(
              false, af[mt], false, bfrag[nt],
              (short)0, acc[mt][nt], false, false);
    }
    __syncthreads();
  }

  // ---- epilogue: C/D layout => VGPR r: M = base + lhi*8 + r, N = base + lm
#pragma unroll
  for (int mt = 0; mt < 2; ++mt) {
    const int row0 = bm + wm + mt * 16 + lhi * 8;
#pragma unroll
    for (int nt = 0; nt < 4; ++nt) {
      const int col = bn + wn + nt * 16 + lm;
      if (col < N) {
#pragma unroll
        for (int r = 0; r < 8; ++r)
          C[(size_t)(row0 + r) * N + col] = acc[mt][nt][r];
      }
    }
  }
}

// ---------------------------------------------------------------------------
// Causal depthwise conv1d (4 taps) + bias + SiLU on the x half of xz.
// ---------------------------------------------------------------------------
__global__ void k_conv_silu(const float* __restrict__ xz,
                            const float* __restrict__ conv_w,
                            const float* __restrict__ conv_b,
                            float* __restrict__ xf,
                            bf16* __restrict__ xb) {
  const size_t total = (size_t)BLM * D_INNER;
  size_t i = (size_t)blockIdx.x * blockDim.x + threadIdx.x;
  if (i >= total) return;
  const int c = (int)(i % D_INNER);
  const int l = (int)((i / D_INNER) % SEQ);
  const int b = (int)(i / ((size_t)D_INNER * SEQ));
  float acc = conv_b[c];
#pragma unroll
  for (int k = 0; k < 4; ++k) {
    const int lt = l + k - 3;
    if (lt >= 0)
      acc += xz[((size_t)(b * SEQ + lt)) * (2 * D_INNER) + c] * conv_w[c * 4 + k];
  }
  const float s = acc / (1.0f + __expf(-acc));
  xf[i] = s;
  xb[i] = (bf16)s;
}

// ---------------------------------------------------------------------------
// dt_r slice (first 128 cols of ssm) -> bf16
// ---------------------------------------------------------------------------
__global__ void k_extract_dtr(const float* __restrict__ ssm,
                              bf16* __restrict__ dtr) {
  const size_t total = (size_t)BLM * DT_RANK;
  size_t i = (size_t)blockIdx.x * blockDim.x + threadIdx.x;
  if (i >= total) return;
  const int    col = (int)(i % DT_RANK);
  const size_t row = i / DT_RANK;
  dtr[i] = (bf16)ssm[row * SSM_N + col];
}

// ---------------------------------------------------------------------------
// delta = softplus(delta_raw + dt_bias)  (in place)
// ---------------------------------------------------------------------------
__global__ void k_softplus_bias(float* __restrict__ delta,
                                const float* __restrict__ dt_bias) {
  const size_t total = (size_t)BLM * D_INNER;
  size_t i = (size_t)blockIdx.x * blockDim.x + threadIdx.x;
  if (i >= total) return;
  const int c = (int)(i % D_INNER);
  const float x = delta[i] + dt_bias[c];
  delta[i] = (x > 20.0f) ? x : log1pf(__expf(x));
}

// ---------------------------------------------------------------------------
// Segmented selective scan.
// Segment transition is affine: h_end = exp(a_n * sum(dt)) * h_start + bacc_n,
// since A is time-invariant. Three passes:
//   A) per-segment (sum_dt, bacc[16])  -- 64x the parallelism of a flat scan
//   B) sequential combine over 64 segments per channel -> h_start per segment
//   C) replay each segment from h_start, emit gated y (bf16)
// ---------------------------------------------------------------------------
__global__ __launch_bounds__(256)
void k_scan_seg(const float* __restrict__ delta,
                const float* __restrict__ xf,
                const float* __restrict__ ssm,
                const float* __restrict__ Amat,
                float* __restrict__ segsum,
                float* __restrict__ segbacc) {
  const int c   = blockIdx.x * blockDim.x + threadIdx.x;
  const int seg = blockIdx.y;
  const int b   = blockIdx.z;

  float a[D_STATE], bacc[D_STATE];
#pragma unroll
  for (int n = 0; n < D_STATE; ++n) { a[n] = Amat[c * D_STATE + n]; bacc[n] = 0.0f; }
  float sumdt = 0.0f;

  const int t0 = seg * SEGLEN;
  for (int t = t0; t < t0 + SEGLEN; ++t) {
    const size_t rc = (size_t)(b * SEQ + t);
    const float dt = delta[rc * D_INNER + c];
    const float u  = xf[rc * D_INNER + c];
    const float* Bv = &ssm[rc * SSM_N + DT_RANK];
    __builtin_prefetch(&ssm[(rc + 1) * SSM_N + DT_RANK], 0, 1); // global_prefetch_b8
    const float du = dt * u;
    sumdt += dt;
#pragma unroll
    for (int n = 0; n < D_STATE; ++n)
      bacc[n] = bacc[n] * __expf(dt * a[n]) + du * Bv[n];
  }
  const size_t base = (size_t)(b * SEGS + seg);
  segsum[base * D_INNER + c] = sumdt;
#pragma unroll
  for (int n = 0; n < D_STATE; ++n)
    segbacc[(base * D_STATE + n) * D_INNER + c] = bacc[n];
}

__global__ __launch_bounds__(256)
void k_scan_combine(const float* __restrict__ segsum,
                    const float* __restrict__ segbacc,
                    const float* __restrict__ Amat,
                    float* __restrict__ hstart) {
  const int c = blockIdx.x * blockDim.x + threadIdx.x;
  const int b = blockIdx.y;
  float a[D_STATE], h[D_STATE];
#pragma unroll
  for (int n = 0; n < D_STATE; ++n) { a[n] = Amat[c * D_STATE + n]; h[n] = 0.0f; }

  for (int seg = 0; seg < SEGS; ++seg) {
    const size_t base = (size_t)(b * SEGS + seg);
#pragma unroll
    for (int n = 0; n < D_STATE; ++n)
      hstart[(base * D_STATE + n) * D_INNER + c] = h[n];
    const float s = segsum[base * D_INNER + c];
#pragma unroll
    for (int n = 0; n < D_STATE; ++n)
      h[n] = h[n] * __expf(a[n] * s) + segbacc[(base * D_STATE + n) * D_INNER + c];
  }
}

__global__ __launch_bounds__(256)
void k_scan_emit(const float* __restrict__ delta,
                 const float* __restrict__ xf,
                 const float* __restrict__ ssm,
                 const float* __restrict__ xz,
                 const float* __restrict__ Amat,
                 const float* __restrict__ Dvec,
                 const float* __restrict__ hstart,
                 bf16* __restrict__ yb) {
  const int c   = blockIdx.x * blockDim.x + threadIdx.x;
  const int seg = blockIdx.y;
  const int b   = blockIdx.z;

  const size_t base = (size_t)(b * SEGS + seg);
  float a[D_STATE], h[D_STATE];
#pragma unroll
  for (int n = 0; n < D_STATE; ++n) {
    a[n] = Amat[c * D_STATE + n];
    h[n] = hstart[(base * D_STATE + n) * D_INNER + c];
  }
  const float dskip = Dvec[c];

  const int t0 = seg * SEGLEN;
  for (int t = t0; t < t0 + SEGLEN; ++t) {
    const size_t rc = (size_t)(b * SEQ + t);
    const float dt = delta[rc * D_INNER + c];
    const float u  = xf[rc * D_INNER + c];
    const float z  = xz[rc * (2 * D_INNER) + D_INNER + c];
    const float* Bv = &ssm[rc * SSM_N + DT_RANK];
    const float* Cv = &ssm[rc * SSM_N + DT_RANK + D_STATE];
    __builtin_prefetch(&ssm[(rc + 1) * SSM_N + DT_RANK], 0, 1);
    const float du = dt * u;
    float y = 0.0f;
#pragma unroll
    for (int n = 0; n < D_STATE; ++n) {
      h[n] = h[n] * __expf(dt * a[n]) + du * Bv[n];
      y += h[n] * Cv[n];
    }
    y += u * dskip;
    const float g = z / (1.0f + __expf(-z));
    yb[rc * D_INNER + c] = (bf16)(y * g);
  }
}

// ---------------------------------------------------------------------------
// Host orchestration
// ---------------------------------------------------------------------------
extern "C" void kernel_launch(void* const* d_in, const int* in_sizes, int n_in,
                              void* d_out, int out_size, void* d_ws, size_t ws_size,
                              hipStream_t stream) {
  (void)in_sizes; (void)n_in; (void)out_size; (void)ws_size;

  const float* hs      = (const float*)d_in[0];
  const float* W_in    = (const float*)d_in[1];
  const float* conv_w  = (const float*)d_in[2];
  const float* conv_b  = (const float*)d_in[3];
  const float* W_x     = (const float*)d_in[4];
  const float* W_dt    = (const float*)d_in[5];
  const float* dt_bias = (const float*)d_in[6];
  const float* Amat    = (const float*)d_in[7];
  const float* Dvec    = (const float*)d_in[8];
  const float* W_out   = (const float*)d_in[9];

  char* ws = (char*)d_ws;
  size_t off = 0;
  auto take = [&](size_t bytes) -> char* {
    char* p = ws + off;
    off += (bytes + 255) & ~(size_t)255;
    return p;
  };

  bf16*  hB    = (bf16*) take((size_t)BLM * D_MODEL * sizeof(bf16));
  bf16*  wInB  = (bf16*) take((size_t)2 * D_INNER * D_MODEL * sizeof(bf16));
  float* xz    = (float*)take((size_t)BLM * 2 * D_INNER * sizeof(float));
  float* xf    = (float*)take((size_t)BLM * D_INNER * sizeof(float));
  bf16*  xb    = (bf16*) take((size_t)BLM * D_INNER * sizeof(bf16));
  bf16*  wXB   = (bf16*) take((size_t)SSM_N * D_INNER * sizeof(bf16));
  float* ssm   = (float*)take((size_t)BLM * SSM_N * sizeof(float));
  bf16*  dtrB  = (bf16*) take((size_t)BLM * DT_RANK * sizeof(bf16));
  bf16*  wDtB  = (bf16*) take((size_t)D_INNER * DT_RANK * sizeof(bf16));
  float* delta = (float*)take((size_t)BLM * D_INNER * sizeof(float));
  bf16*  yB    = (bf16*) take((size_t)BLM * D_INNER * sizeof(bf16));
  bf16*  wOutB = (bf16*) take((size_t)D_MODEL * D_INNER * sizeof(bf16));
  float* segsum  = (float*)take((size_t)B_SZ * SEGS * D_INNER * sizeof(float));
  float* segbacc = (float*)take((size_t)B_SZ * SEGS * D_STATE * D_INNER * sizeof(float));
  float* hstart  = (float*)take((size_t)B_SZ * SEGS * D_STATE * D_INNER * sizeof(float));

  auto cast = [&](const float* s, bf16* d, size_t n) {
    k_cast_bf16<<<(unsigned)((n + 255) / 256), 256, 0, stream>>>(s, d, n);
  };
  auto gemm = [&](const bf16* a, const bf16* b, float* c, int M, int N, int K) {
    dim3 grid((unsigned)((N + 127) / 128), (unsigned)(M / 128));
    k_gemm_bf16_wmma<<<grid, 256, 0, stream>>>(a, b, c, M, N, K);
  };

  // 0) bf16 copies
  cast(hs,    hB,    (size_t)BLM * D_MODEL);
  cast(W_in,  wInB,  (size_t)2 * D_INNER * D_MODEL);
  cast(W_x,   wXB,   (size_t)SSM_N * D_INNER);
  cast(W_dt,  wDtB,  (size_t)D_INNER * DT_RANK);
  cast(W_out, wOutB, (size_t)D_MODEL * D_INNER);

  // 1) in_proj
  gemm(hB, wInB, xz, BLM, 2 * D_INNER, D_MODEL);

  // 2) conv + SiLU
  {
    const size_t total = (size_t)BLM * D_INNER;
    k_conv_silu<<<(unsigned)((total + 255) / 256), 256, 0, stream>>>(
        xz, conv_w, conv_b, xf, xb);
  }

  // 3) x_proj (N=160, guarded path)
  gemm(xb, wXB, ssm, BLM, SSM_N, D_INNER);

  // 4) dt_r -> bf16
  {
    const size_t total = (size_t)BLM * DT_RANK;
    k_extract_dtr<<<(unsigned)((total + 255) / 256), 256, 0, stream>>>(ssm, dtrB);
  }

  // 5) dt_proj
  gemm(dtrB, wDtB, delta, BLM, D_INNER, DT_RANK);

  // 6) softplus
  {
    const size_t total = (size_t)BLM * D_INNER;
    k_softplus_bias<<<(unsigned)((total + 255) / 256), 256, 0, stream>>>(
        delta, dt_bias);
  }

  // 7) segmented selective scan
  {
    dim3 gridA(D_INNER / 256, SEGS, B_SZ);
    k_scan_seg<<<gridA, 256, 0, stream>>>(delta, xf, ssm, Amat, segsum, segbacc);
    dim3 gridB(D_INNER / 256, B_SZ);
    k_scan_combine<<<gridB, 256, 0, stream>>>(segsum, segbacc, Amat, hstart);
    k_scan_emit<<<gridA, 256, 0, stream>>>(delta, xf, ssm, xz, Amat, Dvec,
                                           hstart, yB);
  }

  // 8) out_proj -> d_out (f32)
  gemm(yB, wOutB, (float*)d_out, BLM, D_MODEL, D_INNER);
}
